// Position_encode_1125281431668
// MI455X (gfx1250) — compile-verified, matrix operand order checked by reading, not run on hardware
//
#include <hip/hip_runtime.h>
#include <math.h>

// ---- types ----
typedef _Float16 half_t;
typedef __attribute__((ext_vector_type(16))) _Float16       v16h;
typedef __attribute__((ext_vector_type(8)))  _Float16       v8h;
typedef __attribute__((ext_vector_type(8)))  float          v8f;
typedef __attribute__((ext_vector_type(8)))  unsigned short v8us;

#define DIMS    64
#define K_POS   32
#define K_NEG   64
#define WAVES   8     // waves per block (256 threads, wave32)

// |x| for packed f16 via sign-bit clear (lowers to packed VALU ops)
static __device__ __forceinline__ v8h habs8(v8h x) {
  v8us u = __builtin_bit_cast(v8us, x);
  u &= (unsigned short)0x7FFF;
  return __builtin_bit_cast(v8h, u);
}

static __device__ __forceinline__ v16h pack16(v8h lo, v8h hi) {
  v16h r;
#pragma unroll
  for (int e = 0; e < 8; ++e) { r[e] = lo[e]; r[e + 8] = hi[e]; }
  return r;
}

// One 16-neighbor group: abs-diff in f16 (VALU, packed) + K-reduction on the
// matrix pipe. A layout matches ISA 16-bit A 16x32: lane L<16 holds
// K={c*32+0..7, c*32+16..23}, lane L+16 holds K={c*32+8..15, c*32+24..31}.
// B = all-ones -> D[m][n] = H_m for every n (layout-invariant).
static __device__ __forceinline__ v8f group_dist(const half_t* __restrict__ Zh,
                                                 int nbr, int sub,
                                                 v8h zi0a, v8h zi0b,
                                                 v8h zi1a, v8h zi1b,
                                                 v16h bone) {
  const v8h* zr = (const v8h*)(Zh + (size_t)nbr * DIMS);
  v8f acc = {};
  // chunk 0: dims 0..31
  v16h A = pack16(habs8(zr[0 + sub] - zi0a), habs8(zr[2 + sub] - zi0b));
  acc = __builtin_amdgcn_wmma_f32_16x16x32_f16(false, A, false, bone,
                                               (short)0, acc, false, false);
  // chunk 1: dims 32..63 (chain accumulator)
  A = pack16(habs8(zr[4 + sub] - zi1a), habs8(zr[6 + sub] - zi1b));
  acc = __builtin_amdgcn_wmma_f32_16x16x32_f16(false, A, false, bone,
                                               (short)0, acc, false, false);
  return acc;  // lane<16: H[0..7] in acc[0..7]; lane>=16: H[8..15]
}

__global__ void init_out_kernel(float* out) {
  if (threadIdx.x < 3) out[threadIdx.x] = 0.0f;
}

// Z = sigmoid(P) -> f16 table; L_deg = mean((Z@W_d - deg)^2). One wave per row.
__global__ void prep_kernel(const float* __restrict__ P,
                            const float* __restrict__ Wd,
                            const float* __restrict__ deg,
                            half_t* __restrict__ Zh,
                            float* __restrict__ out, int N) {
  const int lane = threadIdx.x & 31;
  const int wave = threadIdx.x >> 5;
  const int i = blockIdx.x * WAVES + wave;
  if (i >= N) return;  // wave-uniform

  const size_t base = (size_t)i * DIMS;
  float p0 = P[base + lane];
  float p1 = P[base + lane + 32];
  float z0 = 1.0f / (1.0f + expf(-p0));
  float z1 = 1.0f / (1.0f + expf(-p1));
  Zh[base + lane]      = (half_t)z0;
  Zh[base + lane + 32] = (half_t)z1;

  float dot = z0 * Wd[lane] + z1 * Wd[lane + 32];
#pragma unroll
  for (int off = 16; off >= 1; off >>= 1)
    dot += __shfl_xor(dot, off, 32);
  if (lane == 0) {
    float e = dot - deg[i];
    atomicAdd(out + 2, e * e / (float)N);
  }
}

// One wave per node; blockIdx.y selects which contrast (adj vs deg_dist).
__global__ void contrast_kernel(const half_t* __restrict__ Zh,
                                const int* __restrict__ posA,
                                const int* __restrict__ negA,
                                const int* __restrict__ posB,
                                const int* __restrict__ negB,
                                float* __restrict__ out, int N) {
  const int which = blockIdx.y;
  const int* __restrict__ pos = which ? posB : posA;
  const int* __restrict__ neg = which ? negB : negA;

  const int lane = threadIdx.x & 31;
  const int wave = threadIdx.x >> 5;
  const int i = blockIdx.x * WAVES + wave;
  if (i >= N) return;  // wave-uniform; EXEC stays all-ones for WMMA

  __shared__ float hn[WAVES][K_NEG];

  const int sub = (lane >= 16) ? 1 : 0;
  const v8h* zi = (const v8h*)(Zh + (size_t)i * DIMS);
  const v8h zi0a = zi[0 + sub], zi0b = zi[2 + sub];
  const v8h zi1a = zi[4 + sub], zi1b = zi[6 + sub];

  v16h bone;
#pragma unroll
  for (int e = 0; e < 16; ++e) bone[e] = (_Float16)1.0f;

  // ---- positives: mean over 32 ----
  float hp_lane = 0.0f;
#pragma unroll
  for (int g = 0; g < K_POS / 16; ++g) {
    int nbr = pos[(size_t)i * K_POS + g * 16 + (lane & 15)];
    v8f h = group_dist(Zh, nbr, sub, zi0a, zi0b, zi1a, zi1b, bone);
#pragma unroll
    for (int r = 0; r < 8; ++r) hp_lane += h[r];
  }
  // lane 0 carries sum(H[0..7]), lane 16 carries sum(H[8..15])
  float hp_tot = __shfl(hp_lane, 0, 32) + __shfl(hp_lane, 16, 32);

  // ---- negatives: logsumexp over 64 ----
#pragma unroll
  for (int g = 0; g < K_NEG / 16; ++g) {
    int nbr = neg[(size_t)i * K_NEG + g * 16 + (lane & 15)];
    v8f h = group_dist(Zh, nbr, sub, zi0a, zi0b, zi1a, zi1b, bone);
    if ((lane & 15) == 0) {
      int basek = g * 16 + (lane >> 4) * 8;  // lane0 -> +0, lane16 -> +8
#pragma unroll
      for (int r = 0; r < 8; ++r) hn[wave][basek + r] = h[r];
    }
  }
  // same-wave DS ordering: wait LDS writes before cross-lane reads
  asm volatile("s_wait_dscnt 0x0" ::: "memory");

  float v0 = hn[wave][lane];
  float v1 = hn[wave][lane + 32];
  float m = fmaxf(v0, v1);
#pragma unroll
  for (int off = 16; off >= 1; off >>= 1)
    m = fmaxf(m, __shfl_xor(m, off, 32));
  float s = expf(v0 - m) + expf(v1 - m);
#pragma unroll
  for (int off = 16; off >= 1; off >>= 1)
    s += __shfl_xor(s, off, 32);
  float lse = m + logf(s);

  // L = sum_i (lse_i - mean(Hp_i))
  if (lane == 0) atomicAdd(out + which, lse - hp_tot * (1.0f / (float)K_POS));
}

extern "C" void kernel_launch(void* const* d_in, const int* in_sizes, int n_in,
                              void* d_out, int out_size, void* d_ws, size_t ws_size,
                              hipStream_t stream) {
  const float* P    = (const float*)d_in[0];
  const float* Wd   = (const float*)d_in[1];
  const float* deg  = (const float*)d_in[2];
  const int*   pos  = (const int*)d_in[3];
  const int*   neg  = (const int*)d_in[4];
  const int*   dpos = (const int*)d_in[5];
  const int*   dneg = (const int*)d_in[6];
  float* out = (float*)d_out;
  const int N = in_sizes[2];  // deg_vec length

  half_t* Zh = (half_t*)d_ws;  // N*64 f16 = 6.4 MB, fits workspace / L2

  init_out_kernel<<<1, 32, 0, stream>>>(out);

  const int blocks = (N + WAVES - 1) / WAVES;
  prep_kernel<<<blocks, WAVES * 32, 0, stream>>>(P, Wd, deg, Zh, out, N);

  dim3 grid(blocks, 2);
  contrast_kernel<<<grid, WAVES * 32, 0, stream>>>(Zh, pos, neg, dpos, dneg,
                                                   out, N);
}
